// GroupedQueryAttention_61787399520457
// MI455X (gfx1250) — compile-verified, matrix operand order checked by reading, not run on hardware
//
#include <hip/hip_runtime.h>

#define HIDDEN 2048
#define NH 16
#define NKV 4
#define HD 128
#define NREP (NH / NKV)
#define BATCH 2
#define SEQ 2048
#define NTOK (BATCH * SEQ)
#define QDIM (NH * HD)    // 2048
#define KVDIM (NKV * HD)  // 512
#define SM_SCALE 0.08838834764831845f  // 128^-0.5
#define NEGINF (-1.0e9f)

typedef __bf16 bf16_t;
typedef __attribute__((ext_vector_type(16))) __bf16 v16bf;
typedef __attribute__((ext_vector_type(8)))  __bf16 v8bf;
typedef __attribute__((ext_vector_type(8)))  float  v8f;

// f32 -> bf16, round-to-nearest-even (no reliance on __bf16 arithmetic).
__device__ __forceinline__ bf16_t f2bf(float f) {
  union { float f; unsigned int u; } c; c.f = f;
  unsigned int u = c.u;
  unsigned int r = (u + 0x7FFFu + ((u >> 16) & 1u)) >> 16;
  union { unsigned short s; bf16_t b; } o; o.s = (unsigned short)r;
  return o.b;
}

// Load one 16(row) x 32(K) bf16 fragment (A-frag layout; B-frags use the same
// per-lane pattern with row == the B-matrix N column's source row in memory).
// ISA 7.12.2: lane&15 = row, lane>>4 selects which 8-element K chunks.
__device__ __forceinline__ v16bf load_frag(const bf16_t* base, int ld) {
  const int lane = threadIdx.x & 31;
  const bf16_t* p = base + (size_t)(lane & 15) * ld + ((lane >> 4) << 3);
  v8bf c0 = *reinterpret_cast<const v8bf*>(p);       // K = k0 .. k0+7
  v8bf c1 = *reinterpret_cast<const v8bf*>(p + 16);  // K = k0+16 .. k0+23
  v16bf f;
#pragma unroll
  for (int i = 0; i < 8; ++i) { f[i] = c0[i]; f[i + 8] = c1[i]; }
  return f;
}

__device__ __forceinline__ v8f wmma_bf16(v16bf a, v16bf b, v8f c) {
  return __builtin_amdgcn_wmma_f32_16x16x32_bf16(
      /*neg_a=*/false, a, /*neg_b=*/false, b,
      /*c_mod=*/(short)0, c, /*reuse_a=*/false, /*reuse_b=*/false);
}

// ---------------------------------------------------------------- convert ---
__global__ __launch_bounds__(256) void k_f32_to_bf16(const float* __restrict__ in,
                                                     bf16_t* __restrict__ out, int n) {
  int i = blockIdx.x * blockDim.x + threadIdx.x;
  int stride = gridDim.x * blockDim.x;
  for (; i < n; i += stride) out[i] = f2bf(in[i]);
}

// -------------------------------------------------------------- bf16 GEMM ---
// C[M,N] (f32, row-major) = A[M,K] (bf16 rm) * Bw[N,K]^T (bf16 rm, weight rows)
// Block: 128x128 tile, 256 threads = 8 waves; wave tile 32x64 = 2x4 WMMA frags.
__global__ __launch_bounds__(256) void k_gemm_bf16(const bf16_t* __restrict__ A,
                                                   const bf16_t* __restrict__ Bw,
                                                   float* __restrict__ C,
                                                   int M, int N, int K) {
  const int wid  = threadIdx.x >> 5;
  const int lane = threadIdx.x & 31;
  const int bm = blockIdx.y * 128 + (wid & 3) * 32;
  const int bn = blockIdx.x * 128 + (wid >> 2) * 64;

  v8f acc[2][4];
#pragma unroll
  for (int i = 0; i < 2; ++i)
#pragma unroll
    for (int j = 0; j < 4; ++j) acc[i][j] = v8f{};

  const bf16_t* Ap = A + (size_t)bm * K;
  const bf16_t* Bp = Bw + (size_t)bn * K;

  for (int kk = 0; kk < K; kk += 32) {
    if (kk + 64 < K) {  // warm L2/L0 for the next K step (global_prefetch)
      __builtin_prefetch(Ap + kk + 64, 0, 1);
      __builtin_prefetch(Bp + kk + 64, 0, 1);
    }
    v16bf a0 = load_frag(Ap + kk, K);
    v16bf a1 = load_frag(Ap + (size_t)16 * K + kk, K);
    v16bf b0 = load_frag(Bp + kk, K);
    v16bf b1 = load_frag(Bp + (size_t)16 * K + kk, K);
    v16bf b2 = load_frag(Bp + (size_t)32 * K + kk, K);
    v16bf b3 = load_frag(Bp + (size_t)48 * K + kk, K);
    acc[0][0] = wmma_bf16(a0, b0, acc[0][0]);
    acc[0][1] = wmma_bf16(a0, b1, acc[0][1]);
    acc[0][2] = wmma_bf16(a0, b2, acc[0][2]);
    acc[0][3] = wmma_bf16(a0, b3, acc[0][3]);
    acc[1][0] = wmma_bf16(a1, b0, acc[1][0]);
    acc[1][1] = wmma_bf16(a1, b1, acc[1][1]);
    acc[1][2] = wmma_bf16(a1, b2, acc[1][2]);
    acc[1][3] = wmma_bf16(a1, b3, acc[1][3]);
  }

  const int nloc  = lane & 15;
  const int mhalf = (lane >> 4) * 8;  // C-frag: row = vgpr + 8*(lane>>4)
#pragma unroll
  for (int i = 0; i < 2; ++i)
#pragma unroll
    for (int j = 0; j < 4; ++j) {
      float* cp = C + (size_t)(bm + i * 16 + mhalf) * N + (bn + j * 16 + nloc);
#pragma unroll
      for (int g = 0; g < 8; ++g) cp[(size_t)g * N] = acc[i][j][g];
    }
}

// ------------------------------------------------------------- RoPE + pack ---
// Per token: rotate Q/K, write new_k/new_v (f32 outputs), pack bf16 copies:
//   Qb: (b, h, s, d)   Kb: (b, kvh, s, d)   Vt: (b, kvh, d, s)  (transposed!)
__global__ __launch_bounds__(256) void k_rope_pack(
    const float* __restrict__ Qf, const float* __restrict__ Kf,
    const float* __restrict__ Vf, const float* __restrict__ freqs,
    bf16_t* __restrict__ Qb, bf16_t* __restrict__ Kb, bf16_t* __restrict__ Vt,
    float* __restrict__ outK, float* __restrict__ outV) {
  const int t = blockIdx.x;           // token = b*SEQ + s
  const int b = t / SEQ, s = t % SEQ;
  const float* fr = freqs + (size_t)s * (HD / 2);

  for (int idx = threadIdx.x; idx < NH * (HD / 2); idx += 256) {
    const int h = idx >> 6, j = idx & 63;
    const float* qp = Qf + (size_t)t * QDIM + h * HD;
    float sn, cs; __sincosf(fr[j], &sn, &cs);
    const float x1 = qp[2 * j], x2 = qp[2 * j + 1];
    bf16_t* qb = Qb + (((size_t)(b * NH + h)) * SEQ + s) * HD;
    qb[2 * j]     = f2bf(x1 * cs - x2 * sn);
    qb[2 * j + 1] = f2bf(x1 * sn + x2 * cs);
  }
  for (int idx = threadIdx.x; idx < NKV * (HD / 2); idx += 256) {
    const int h = idx >> 6, j = idx & 63;
    const float* kp = Kf + (size_t)t * KVDIM + h * HD;
    float sn, cs; __sincosf(fr[j], &sn, &cs);
    const float x1 = kp[2 * j], x2 = kp[2 * j + 1];
    const float r1 = x1 * cs - x2 * sn, r2 = x1 * sn + x2 * cs;
    float* ok = outK + (size_t)t * KVDIM + h * HD;
    ok[2 * j] = r1; ok[2 * j + 1] = r2;
    bf16_t* kb = Kb + (((size_t)(b * NKV + h)) * SEQ + s) * HD;
    kb[2 * j] = f2bf(r1); kb[2 * j + 1] = f2bf(r2);
  }
  for (int idx = threadIdx.x; idx < KVDIM; idx += 256) {
    const float v = Vf[(size_t)t * KVDIM + idx];
    outV[(size_t)t * KVDIM + idx] = v;
    const int h = idx >> 7, d = idx & 127;
    Vt[(((size_t)(b * NKV + h)) * HD + d) * SEQ + s] = f2bf(v);
  }
}

// --------------------------------------------------------- flash attention ---
// Grid: (SEQ/128, NH, BATCH). 8 waves; wave w owns q rows [tile+16w, tile+16w+16).
// Per 32-key block: 8 WMMA (QK^T) + online softmax + LDS P relayout + 8 WMMA (PV).
__global__ __launch_bounds__(256) void k_attention(const bf16_t* __restrict__ Qb,
                                                   const bf16_t* __restrict__ Kb,
                                                   const bf16_t* __restrict__ Vt,
                                                   bf16_t* __restrict__ Ob) {
  __shared__ __align__(16) bf16_t p_lds[8][16 * 32];  // per-wave P scratch
  const int wid  = threadIdx.x >> 5;
  const int lane = threadIdx.x & 31;
  const int b = blockIdx.z, h = blockIdx.y, kvh = h / NREP;
  const int q0 = blockIdx.x * 128 + wid * 16;

  const bf16_t* Qp = Qb + (((size_t)(b * NH + h)) * SEQ + q0) * HD;
  const bf16_t* Kp = Kb + ((size_t)(b * NKV + kvh)) * SEQ * HD;
  const bf16_t* Vp = Vt + ((size_t)(b * NKV + kvh)) * HD * SEQ;

  v16bf qa[4];
#pragma unroll
  for (int c = 0; c < 4; ++c) qa[c] = load_frag(Qp + c * 32, HD);

  v8f o[8];
#pragma unroll
  for (int f = 0; f < 8; ++f) o[f] = v8f{};
  float mi[8], li[8];
#pragma unroll
  for (int g = 0; g < 8; ++g) { mi[g] = -3.0e38f; li[g] = 0.0f; }

  const int nloc  = lane & 15;
  const int mrow0 = q0 + ((lane >> 4) << 3);
  const int q_hi  = q0 + 15;  // causal: never visit keys past our last row

  for (int kb = 0; kb <= q_hi; kb += 32) {
    v8f sA = v8f{}, sB = v8f{};
#pragma unroll
    for (int c = 0; c < 4; ++c) {
      v16bf kfA = load_frag(Kp + (size_t)kb * HD + c * 32, HD);
      v16bf kfB = load_frag(Kp + (size_t)(kb + 16) * HD + c * 32, HD);
      sA = wmma_bf16(qa[c], kfA, sA);
      sB = wmma_bf16(qa[c], kfB, sB);
    }
    const int nA = kb + nloc, nB = nA + 16;
#pragma unroll
    for (int g = 0; g < 8; ++g) {
      const int mrow = mrow0 + g;
      float a  = sA[g] * SM_SCALE + (nA <= mrow ? 0.0f : NEGINF);
      float bb = sB[g] * SM_SCALE + (nB <= mrow ? 0.0f : NEGINF);
      // row reduction inside each 16-lane half (wave32: xor 1,2,4,8)
      float rm = fmaxf(a, bb);
      rm = fmaxf(rm, __shfl_xor(rm, 1));
      rm = fmaxf(rm, __shfl_xor(rm, 2));
      rm = fmaxf(rm, __shfl_xor(rm, 4));
      rm = fmaxf(rm, __shfl_xor(rm, 8));
      const float mnew  = fmaxf(mi[g], rm);
      const float alpha = __expf(mi[g] - mnew);
      const float pa = __expf(a - mnew);
      const float pb = __expf(bb - mnew);
      float rs = pa + pb;
      rs += __shfl_xor(rs, 1);
      rs += __shfl_xor(rs, 2);
      rs += __shfl_xor(rs, 4);
      rs += __shfl_xor(rs, 8);
      li[g] = li[g] * alpha + rs;
      mi[g] = mnew;
#pragma unroll
      for (int f = 0; f < 8; ++f) o[f][g] *= alpha;
      // C-layout -> row-major 16x32 P tile in LDS (bf16)
      const int mloc = ((lane >> 4) << 3) + g;
      p_lds[wid][mloc * 32 + nloc]      = f2bf(pa);
      p_lds[wid][mloc * 32 + 16 + nloc] = f2bf(pb);
    }
    asm volatile("s_wait_dscnt 0" ::: "memory");  // cross-lane LDS RAW fence
    v16bf pf = load_frag(&p_lds[wid][0], 32);     // P as A-fragment
#pragma unroll
    for (int f = 0; f < 8; ++f) {
      // B-frag of V (32 keys x 16 dims): rows of Vt, contiguous over keys
      v16bf vf = load_frag(Vp + (size_t)(f * 16) * SEQ + kb, SEQ);
      o[f] = wmma_bf16(pf, vf, o[f]);
    }
    asm volatile("s_wait_dscnt 0" ::: "memory");  // drain before next overwrite
  }

#pragma unroll
  for (int f = 0; f < 8; ++f)
#pragma unroll
    for (int g = 0; g < 8; ++g) {
      const int m = mrow0 + g;
      Ob[((size_t)(b * SEQ + m)) * QDIM + h * HD + f * 16 + nloc] =
          f2bf(o[f][g] / li[g]);
    }
}

// ----------------------------------------------------------------- launch ---
extern "C" void kernel_launch(void* const* d_in, const int* in_sizes, int n_in,
                              void* d_out, int out_size, void* d_ws, size_t ws_size,
                              hipStream_t stream) {
  const float* x    = (const float*)d_in[0];
  const float* wq   = (const float*)d_in[1];
  const float* wk   = (const float*)d_in[2];
  const float* wv   = (const float*)d_in[3];
  const float* wo   = (const float*)d_in[4];
  const float* rope = (const float*)d_in[5];
  // d_in[6] (mask) unused: causal mask computed analytically.

  float* out  = (float*)d_out;
  float* outK = out + (size_t)NTOK * HIDDEN;
  float* outV = outK + (size_t)NTOK * KVDIM;

  char* ws = (char*)d_ws;
  size_t off = 0;
  auto alloc = [&](size_t bytes) {
    char* p = ws + off; off += (bytes + 255) & ~(size_t)255; return (void*)p;
  };
  bf16_t* xb  = (bf16_t*)alloc((size_t)NTOK * HIDDEN * 2);
  bf16_t* wqb = (bf16_t*)alloc((size_t)QDIM * HIDDEN * 2);
  bf16_t* wkb = (bf16_t*)alloc((size_t)KVDIM * HIDDEN * 2);
  bf16_t* wvb = (bf16_t*)alloc((size_t)KVDIM * HIDDEN * 2);
  bf16_t* wob = (bf16_t*)alloc((size_t)HIDDEN * QDIM * 2);
  float*  Qf  = (float*)alloc((size_t)NTOK * QDIM * 4);
  float*  Kf  = (float*)alloc((size_t)NTOK * KVDIM * 4);
  float*  Vf  = (float*)alloc((size_t)NTOK * KVDIM * 4);
  bf16_t* Qb  = (bf16_t*)alloc((size_t)NTOK * QDIM * 2);
  bf16_t* Kb  = (bf16_t*)alloc((size_t)NTOK * KVDIM * 2);
  bf16_t* Vt  = (bf16_t*)alloc((size_t)NTOK * KVDIM * 2);
  bf16_t* Ob  = (bf16_t*)alloc((size_t)NTOK * QDIM * 2);

  auto cvt = [&](const float* src, bf16_t* dst, int n) {
    int blocks = (n + 256 * 8 - 1) / (256 * 8);
    k_f32_to_bf16<<<blocks, 256, 0, stream>>>(src, dst, n);
  };
  cvt(x,  xb,  NTOK * HIDDEN);
  cvt(wq, wqb, QDIM * HIDDEN);
  cvt(wk, wkb, KVDIM * HIDDEN);
  cvt(wv, wvb, KVDIM * HIDDEN);
  cvt(wo, wob, HIDDEN * QDIM);

  // QKV projections (weights L2-resident: 41 MB << 192 MB L2)
  k_gemm_bf16<<<dim3(QDIM / 128, NTOK / 128), 256, 0, stream>>>(
      xb, wqb, Qf, NTOK, QDIM, HIDDEN);
  k_gemm_bf16<<<dim3(KVDIM / 128, NTOK / 128), 256, 0, stream>>>(
      xb, wkb, Kf, NTOK, KVDIM, HIDDEN);
  k_gemm_bf16<<<dim3(KVDIM / 128, NTOK / 128), 256, 0, stream>>>(
      xb, wvb, Vf, NTOK, KVDIM, HIDDEN);

  k_rope_pack<<<NTOK, 256, 0, stream>>>(Qf, Kf, Vf, rope, Qb, Kb, Vt, outK, outV);

  k_attention<<<dim3(SEQ / 128, NH, BATCH), 256, 0, stream>>>(Qb, Kb, Vt, Ob);

  k_gemm_bf16<<<dim3(HIDDEN / 128, NTOK / 128), 256, 0, stream>>>(
      Ob, wob, out, NTOK, HIDDEN, HIDDEN);
}
